// MultiHeadAttention_6837587936035
// MI455X (gfx1250) — matrix operand order repaired and checked by reading, not compiled, not run on hardware
//
#include <hip/hip_runtime.h>
#include <hip/hip_bf16.h>

// Shapes (fixed by the reference)
#define BB   4
#define SS   2048
#define DIN  1024
#define DOUT 1024
#define NH   16
#define HDIM 64
#define NTOK (BB * SS)   // 8192

// LDS A-strip row padding (elements): +8 bf16 => row stride 2064B = 516 dwords,
// 516 % 64 = 4 -> the 16 fragment rows map to distinct LDS bank groups.
#define APAD 8

typedef __attribute__((ext_vector_type(16))) __bf16 bf16x16;
typedef __attribute__((ext_vector_type(8)))  __bf16 bf16x8;
typedef __attribute__((ext_vector_type(8)))  float  f32x8;

// exact parameter type of the async-to-LDS builtin (per hipcc diagnostic)
typedef int int4v __attribute__((vector_size(16)));
typedef __attribute__((address_space(1))) int4v g_int4v;
typedef __attribute__((address_space(3))) int4v l_int4v;

// ---- gfx1250 async-to-LDS path (guarded; falls back to plain LDS stores) ----
#if defined(__gfx1250__) && __has_builtin(__builtin_amdgcn_global_load_async_to_lds_b128)
#define HAVE_ASYNC_LDS 1
#endif

#if __has_builtin(__builtin_amdgcn_s_wait_asynccnt)
#define ASYNC_WAIT0() __builtin_amdgcn_s_wait_asynccnt(0)
#else
#define ASYNC_WAIT0() asm volatile("s_wait_asynccnt 0" ::: "memory")
#endif

static __device__ __forceinline__ bf16x16 cat8(bf16x8 lo, bf16x8 hi) {
    bf16x16 r;
#pragma unroll
    for (int i = 0; i < 8; ++i) { r[i] = lo[i]; r[i + 8] = hi[i]; }
    return r;
}

static __device__ __forceinline__ f32x8 wmma_bf16(bf16x16 a, bf16x16 b, f32x8 c) {
    return __builtin_amdgcn_wmma_f32_16x16x32_bf16(false, a, false, b, (short)0, c,
                                                   false, false);
}

// Stage a 32-row x DIN activation strip (rows t0..t0+31 of src) into LDS,
// cooperatively by 256 threads; 16B chunks, async when available.
static __device__ __forceinline__ void stage_A(const __bf16* __restrict__ src_strip,
                                               __bf16 (*As)[DIN + APAD]) {
#pragma unroll
    for (int i = 0; i < 16; ++i) {
        const int c = threadIdx.x + i * 256;     // 4096 chunks of 16B
        const int r = c >> 7;                    // 128 chunks per 1024-elem row
        const int k = (c & 127) * 8;
#if defined(HAVE_ASYNC_LDS)
        __builtin_amdgcn_global_load_async_to_lds_b128(
            (g_int4v*)(src_strip + (size_t)r * DIN + k),
            (l_int4v*)(&As[r][k]), 0, 0);
#else
        *(uint4*)(&As[r][k]) = *(const uint4*)(src_strip + (size_t)r * DIN + k);
#endif
    }
#if defined(HAVE_ASYNC_LDS)
    ASYNC_WAIT0();
#endif
    __syncthreads();
}

// ---------------------------------------------------------------- converts ---
__global__ void cvt_bf16_x4(const float* __restrict__ in, __bf16* __restrict__ out, int n) {
    int i = (blockIdx.x * blockDim.x + threadIdx.x) * 4;
    if (i + 3 < n) {
        float4 v = *(const float4*)(in + i);
        out[i + 0] = (__bf16)v.x;
        out[i + 1] = (__bf16)v.y;
        out[i + 2] = (__bf16)v.z;
        out[i + 3] = (__bf16)v.w;
    }
}

// wt[j][i] = (bf16) w[i][j]  for a 1024x1024 weight (coalesced writes)
__global__ void transpose_cvt(const float* __restrict__ w, __bf16* __restrict__ wt) {
    int idx = blockIdx.x * blockDim.x + threadIdx.x;   // 0..1M-1
    int j = idx >> 10;
    int i = idx & 1023;
    wt[idx] = (__bf16)w[i * DOUT + j];
}

// -------------------------------------------------------------- QKV GEMM ----
// Block: 256 threads (8 waves), shared 32-token A strip in LDS.
// Wave tile: M=32 (two 16-row A tiles) x N=64 -> 8 WMMA per k-step,
// 4 global B loads + 4 LDS A loads per k-step (0.5 global loads / WMMA).
// blockIdx.z selects Q/K/V. Q gets 1/sqrt(HD)=0.125 folded in; V stored
// transposed [b,h,hd,s] so attention's P@V B-fragments are contiguous.
__global__ __launch_bounds__(256) void qkv_gemm(
    const __bf16* __restrict__ xb,
    const __bf16* __restrict__ wqt, const __bf16* __restrict__ wkt,
    const __bf16* __restrict__ wvt,
    __bf16* __restrict__ Qm, __bf16* __restrict__ Km, __bf16* __restrict__ Vtm) {
    __shared__ alignas(16) __bf16 As[32][DIN + APAD];   // ~66 KB

    const int lane = threadIdx.x & 31;
    const int wv   = threadIdx.x >> 5;
    const int hl   = lane >> 4;
    const int lm   = lane & 15;
    const int mode = blockIdx.z;      // 0=Q 1=K 2=V
    const __bf16* wt = (mode == 0) ? wqt : (mode == 1 ? wkt : wvt);
    const int t0 = blockIdx.y * 32;               // token strip base
    const int n0 = blockIdx.x * 512 + wv * 64;    // feature tile base

    stage_A(xb + (size_t)t0 * DIN, As);

    f32x8 acc[2][4] = {{{}, {}, {}, {}}, {{}, {}, {}, {}}};

    for (int k0 = 0; k0 < DIN; k0 += 32) {
        const __bf16* ar0 = &As[lm][k0];
        const __bf16* ar1 = &As[16 + lm][k0];
        bf16x16 a0 = cat8(*(const bf16x8*)(ar0 + hl * 16),
                          *(const bf16x8*)(ar0 + 8 + hl * 16));
        bf16x16 a1 = cat8(*(const bf16x8*)(ar1 + hl * 16),
                          *(const bf16x8*)(ar1 + 8 + hl * 16));
#pragma unroll
        for (int nt = 0; nt < 4; ++nt) {
            const __bf16* wrow = wt + (size_t)(n0 + nt * 16 + lm) * DIN + k0 + hl * 16;
            bf16x16 bfr = *(const bf16x16*)wrow;
            acc[0][nt] = wmma_bf16(a0, bfr, acc[0][nt]);
            acc[1][nt] = wmma_bf16(a1, bfr, acc[1][nt]);
        }
    }

#pragma unroll
    for (int mt = 0; mt < 2; ++mt) {
#pragma unroll
        for (int nt = 0; nt < 4; ++nt) {
            const int dg = n0 + nt * 16 + lm;
            const int h  = dg >> 6;
            const int hd = dg & 63;
#pragma unroll
            for (int e = 0; e < 8; ++e) {
                const int tok = t0 + mt * 16 + e + hl * 8;
                const int b   = tok >> 11;           // /S
                const int sl  = tok & (SS - 1);
                const float v = acc[mt][nt][e];
                if (mode == 0)
                    Qm[((size_t)(b * NH + h) * SS + sl) * HDIM + hd] = (__bf16)(v * 0.125f);
                else if (mode == 1)
                    Km[((size_t)(b * NH + h) * SS + sl) * HDIM + hd] = (__bf16)v;
                else
                    Vtm[((size_t)(b * NH + h) * HDIM + hd) * SS + sl] = (__bf16)v;
            }
        }
    }
}

// ------------------------------------------------------- flash attention ----
// Block = (b, h, 64 q-rows); 4 waves, each owns a 16-row q tile.
// Key blocks of 32 (causally bounded). Online softmax in fp32.
__global__ __launch_bounds__(128) void attn_fwd(
    const __bf16* __restrict__ Qm, const __bf16* __restrict__ Km,
    const __bf16* __restrict__ Vtm, __bf16* __restrict__ ctx) {
    __shared__ alignas(16) __bf16 pst[4][16][40];   // per-wave P staging (+pad)

    const int lane = threadIdx.x & 31;
    const int wv   = threadIdx.x >> 5;
    const int hl   = lane >> 4;
    const int lm   = lane & 15;
    const int h    = blockIdx.y;
    const int b    = blockIdx.z;
    const int bh   = b * NH + h;
    const int q0   = blockIdx.x * 64 + wv * 16;

    const __bf16* Qb = Qm + (size_t)bh * SS * HDIM;
    const __bf16* Kb = Km + (size_t)bh * SS * HDIM;
    const __bf16* Vb = Vtm + (size_t)bh * HDIM * SS;

    // Q A-fragments for d=0..31 and d=32..63 (Q already carries the 1/8 scale)
    bf16x16 qa0, qa1;
    {
        const __bf16* qrow = Qb + (size_t)(q0 + lm) * HDIM;
        qa0 = cat8(*(const bf16x8*)(qrow + hl * 16),
                   *(const bf16x8*)(qrow + 8 + hl * 16));
        qa1 = cat8(*(const bf16x8*)(qrow + 32 + hl * 16),
                   *(const bf16x8*)(qrow + 40 + hl * 16));
    }

    f32x8 o0 = {}, o1 = {}, o2 = {}, o3 = {};
    float mrow[8], lrow[8];
#pragma unroll
    for (int e = 0; e < 8; ++e) { mrow[e] = -3.0e38f; lrow[e] = 0.0f; }

    const int kbmax = (q0 + 15) >> 5;   // last key block intersecting causal region
    for (int kb = 0; kb <= kbmax; ++kb) {
        const int k0 = kb * 32;

        // ---- scores: two 16x16 tiles over d=64 (2 WMMA each) ----
        f32x8 sc0, sc1;
        {
            const __bf16* kr0 = Kb + (size_t)(k0 + lm) * HDIM;
            const __bf16* kr1 = Kb + (size_t)(k0 + 16 + lm) * HDIM;
            f32x8 z0 = {}, z1 = {};
            sc0 = wmma_bf16(qa0, *(const bf16x16*)(kr0 + hl * 16), z0);
            sc0 = wmma_bf16(qa1, *(const bf16x16*)(kr0 + 32 + hl * 16), sc0);
            sc1 = wmma_bf16(qa0, *(const bf16x16*)(kr1 + hl * 16), z1);
            sc1 = wmma_bf16(qa1, *(const bf16x16*)(kr1 + 32 + hl * 16), sc1);
        }

        // ---- causal mask + online softmax (row m = e + hl*8, col = lm) ----
#pragma unroll
        for (int e = 0; e < 8; ++e) {
            const int qg = q0 + e + hl * 8;
            float s0 = sc0[e]; if (k0 + lm      > qg) s0 = -3.0e38f;
            float s1 = sc1[e]; if (k0 + 16 + lm > qg) s1 = -3.0e38f;
            float rm = fmaxf(s0, s1);
#pragma unroll
            for (int off = 8; off >= 1; off >>= 1) rm = fmaxf(rm, __shfl_xor(rm, off));
            const float mn    = fmaxf(mrow[e], rm);
            const float alpha = __expf(mrow[e] - mn);
            mrow[e] = mn;
            const float p0 = __expf(s0 - mn);
            const float p1 = __expf(s1 - mn);
            float rs = p0 + p1;
#pragma unroll
            for (int off = 8; off >= 1; off >>= 1) rs += __shfl_xor(rs, off);
            lrow[e] = lrow[e] * alpha + rs;
            o0[e] *= alpha; o1[e] *= alpha; o2[e] *= alpha; o3[e] *= alpha;
            pst[wv][e + hl * 8][lm]      = (__bf16)p0;
            pst[wv][e + hl * 8][16 + lm] = (__bf16)p1;
        }
        // wave-private LDS region; only need our own DS stores complete
        asm volatile("s_wait_dscnt 0" ::: "memory");

        // ---- re-read P in A-fragment layout ----
        bf16x16 pa;
        {
            const __bf16* prow = &pst[wv][lm][0];
            pa = cat8(*(const bf16x8*)(prow + hl * 16),
                      *(const bf16x8*)(prow + 8 + hl * 16));
        }

        // ---- O += P @ V  (V stored transposed: rows are head dims) ----
        o0 = wmma_bf16(pa, *(const bf16x16*)(Vb + (size_t)(lm)      * SS + k0 + hl * 16), o0);
        o1 = wmma_bf16(pa, *(const bf16x16*)(Vb + (size_t)(16 + lm) * SS + k0 + hl * 16), o1);
        o2 = wmma_bf16(pa, *(const bf16x16*)(Vb + (size_t)(32 + lm) * SS + k0 + hl * 16), o2);
        o3 = wmma_bf16(pa, *(const bf16x16*)(Vb + (size_t)(48 + lm) * SS + k0 + hl * 16), o3);
    }

    // ---- normalize, store ctx as [b, s, h*HD + d] (ready for out-proj) ----
#pragma unroll
    for (int e = 0; e < 8; ++e) {
        const float inv = 1.0f / lrow[e];
        const int sl = q0 + e + hl * 8;
        __bf16* crow = ctx + ((size_t)(b * SS + sl)) * DOUT + h * HDIM;
        crow[lm]      = (__bf16)(o0[e] * inv);
        crow[16 + lm] = (__bf16)(o1[e] * inv);
        crow[32 + lm] = (__bf16)(o2[e] * inv);
        crow[48 + lm] = (__bf16)(o3[e] * inv);
    }
}

// ------------------------------------------------------- output projection --
__global__ __launch_bounds__(256) void out_proj(
    const __bf16* __restrict__ cb, const __bf16* __restrict__ wot,
    const float* __restrict__ bo, float* __restrict__ out) {
    __shared__ alignas(16) __bf16 As[32][DIN + APAD];

    const int lane = threadIdx.x & 31;
    const int wv   = threadIdx.x >> 5;
    const int hl   = lane >> 4;
    const int lm   = lane & 15;
    const int t0 = blockIdx.y * 32;
    const int n0 = blockIdx.x * 512 + wv * 64;

    stage_A(cb + (size_t)t0 * DOUT, As);

    f32x8 acc[2][4] = {{{}, {}, {}, {}}, {{}, {}, {}, {}}};

    for (int k0 = 0; k0 < DOUT; k0 += 32) {
        const __bf16* ar0 = &As[lm][k0];
        const __bf16* ar1 = &As[16 + lm][k0];
        bf16x16 a0 = cat8(*(const bf16x8*)(ar0 + hl * 16),
                          *(const bf16x8*)(ar0 + 8 + hl * 16));
        bf16x16 a1 = cat8(*(const bf16x8*)(ar1 + hl * 16),
                          *(const bf16x8*)(ar1 + 8 + hl * 16));
#pragma unroll
        for (int nt = 0; nt < 4; ++nt) {
            const __bf16* wrow = wot + (size_t)(n0 + nt * 16 + lm) * DOUT + k0 + hl * 16;
            bf16x16 bfr = *(const bf16x16*)wrow;
            acc[0][nt] = wmma_bf16(a0, bfr, acc[0][nt]);
            acc[1][nt] = wmma_bf16(a1, bfr, acc[1][nt]);
        }
    }

#pragma unroll
    for (int mt = 0; mt < 2; ++mt) {
#pragma unroll
        for (int nt = 0; nt < 4; ++nt) {
            const int dg = n0 + nt * 16 + lm;
            const float bias = bo[dg];
#pragma unroll
            for (int e = 0; e < 8; ++e) {
                const int tok = t0 + mt * 16 + e + hl * 8;
                out[(size_t)tok * DOUT + dg] = acc[mt][nt][e] + bias;
            }
        }
    }
}

// ------------------------------------------------------------------ launch --
extern "C" void kernel_launch(void* const* d_in, const int* in_sizes, int n_in,
                              void* d_out, int out_size, void* d_ws, size_t ws_size,
                              hipStream_t stream) {
    const float* x  = (const float*)d_in[0];
    const float* Wq = (const float*)d_in[1];
    const float* Wk = (const float*)d_in[2];
    const float* Wv = (const float*)d_in[3];
    const float* Wo = (const float*)d_in[4];
    const float* bo = (const float*)d_in[5];
    float* out = (float*)d_out;

    // workspace carve (bf16), total ~88 MB
    const size_t NX = (size_t)NTOK * DIN;            // 8M
    const size_t NW = (size_t)DIN * DOUT;            // 1M
    const size_t NQ = (size_t)BB * NH * SS * HDIM;   // 8M
    __bf16* p    = (__bf16*)d_ws;
    __bf16* xb   = p; p += NX;
    __bf16* wqt  = p; p += NW;
    __bf16* wkt  = p; p += NW;
    __bf16* wvt  = p; p += NW;
    __bf16* wot  = p; p += NW;
    __bf16* Qm   = p; p += NQ;
    __bf16* Km   = p; p += NQ;
    __bf16* Vtm  = p; p += NQ;
    __bf16* ctxb = p;

    cvt_bf16_x4<<<(int)(NX / 4 / 256), 256, 0, stream>>>(x, xb, (int)NX);
    transpose_cvt<<<(int)(NW / 256), 256, 0, stream>>>(Wq, wqt);
    transpose_cvt<<<(int)(NW / 256), 256, 0, stream>>>(Wk, wkt);
    transpose_cvt<<<(int)(NW / 256), 256, 0, stream>>>(Wv, wvt);
    transpose_cvt<<<(int)(NW / 256), 256, 0, stream>>>(Wo, wot);

    qkv_gemm<<<dim3(DOUT / 512, NTOK / 32, 3), 256, 0, stream>>>(
        xb, wqt, wkt, wvt, Qm, Km, Vtm);

    attn_fwd<<<dim3(SS / 64, NH, BB), 128, 0, stream>>>(Qm, Km, Vtm, ctxb);

    out_proj<<<dim3(DOUT / 512, NTOK / 32, 1), 256, 0, stream>>>(ctxb, wot, bo, out);
}
